// DCRNNModel_nextTimePred_44074954392028
// MI455X (gfx1250) — matrix operand order repaired and checked by reading, not compiled, not run on hardware
//
#include <hip/hip_runtime.h>

typedef __attribute__((ext_vector_type(16))) __bf16 v16bf;
typedef __attribute__((ext_vector_type(8)))  float  v8f;
typedef __attribute__((ext_vector_type(4)))  unsigned int uint4v;

#define NNODE 64
#define HD 64
#define NTHREADS 256
#define NWAVES 8

// ---- WMMA helpers (contiguous K order, same for A and B, wave32) ----

__device__ __forceinline__ v8f wmma_bf16(v16bf a, v16bf b, v8f c) {
    return __builtin_amdgcn_wmma_f32_16x16x32_bf16(false, a, false, b, (short)0, c, false, false);
}

// A matrix 16x32 bf16 from row-major source (row stride ld). Per-lane: row m0+ln15,
// 16 contiguous K values at k0+half*16. (LDS or global.)
__device__ __forceinline__ v16bf ld_a(const __bf16* base, int ld, int m0, int k0, int lane) {
    const int m  = m0 + (lane & 15);
    const int kb = k0 + ((lane >> 4) << 4);
    const __bf16* p = base + (size_t)m * ld + kb;
    v16bf a;
#pragma unroll
    for (int e = 0; e < 16; ++e) a[e] = p[e];
    return a;
}

// B matrix 32x16 bf16 where B = srcT and src is ROW-major with row stride ld:
// B[K][N] = src[N][K]  -> lane holds column N = n0+ln15, 16 contiguous K values. (LDS or global.)
__device__ __forceinline__ v16bf ld_bt(const __bf16* base, int ld, int k0, int n0, int lane) {
    const int n  = n0 + (lane & 15);
    const int kb = k0 + ((lane >> 4) << 4);
    const __bf16* p = base + (size_t)n * ld + kb;
    v16bf b;
#pragma unroll
    for (int e = 0; e < 16; ++e) b[e] = p[e];
    return b;
}

__device__ __forceinline__ unsigned int pack2bf(float a, float b) {
    union { __bf16 h; unsigned short u; } ca, cb;
    ca.h = (__bf16)a; cb.h = (__bf16)b;
    return (unsigned int)ca.u | ((unsigned int)cb.u << 16);
}

// ---- One DCGRU cell. DIN = 1 (x from sXV) or 64 (x from xmat fp32 64x64). ----
template <int DIN>
__device__ void dcgru_cell(const __bf16* __restrict__ sS,     // 64x64 row-major
                           __bf16* __restrict__ sXC,          // node-major [64][Kp]
                           __bf16* __restrict__ sXCT,         // feat-major [2*Fp][64]
                           float* __restrict__ sRU,
                           const float* __restrict__ sXV,
                           float* __restrict__ hbuf,
                           const float* __restrict__ xmat,
                           const __bf16* __restrict__ Wg, const float* __restrict__ bg,
                           const __bf16* __restrict__ Wc, const float* __restrict__ bc) {
    constexpr int Fp = (DIN == 1) ? 96 : 128;   // per-term K padded to mult of 32
    constexpr int Kp = 3 * Fp;
    constexpr int FT = Fp / 16;                 // feature tiles

    const int tid  = threadIdx.x;
    const int lane = tid & 31;
    const int wv   = tid >> 5;
    const int half = lane >> 4;
    const int ln15 = lane & 15;

    auto x0val = [&](int n, int f, int phase) -> float {
        if (f < DIN) return (DIN == 1) ? sXV[n] : xmat[n * HD + f];
        int j = f - DIN;
        if (j < HD) {
            float hv = hbuf[n * HD + j];
            return phase ? sRU[n * 2 * HD + j] * hv : hv;   // candidate phase: r*h
        }
        return 0.0f;
    };

    for (int phase = 0; phase < 2; ++phase) {   // 0 = gate (r,u), 1 = candidate
        // ---- build x0 into XC block0 (node-major) and XCT block0 (feat-major) ----
        for (int idx = tid; idx < NNODE * Fp; idx += NTHREADS) {
            int n = idx / Fp, f = idx - n * Fp;
            __bf16 v = (__bf16)x0val(n, f, phase);
            sXC[n * Kp + f]    = v;
            sXCT[f * NNODE + n] = v;
        }
        __syncthreads();

        // ---- stage1: x1^T = x0^T @ S^T  -> XC block1 (packed b128) + XCT block1 ----
        for (int t = wv; t < FT * 4; t += NWAVES) {
            int mt = t >> 2, nt = t & 3;
            v8f acc = {};
#pragma unroll
            for (int k0 = 0; k0 < NNODE; k0 += 32) {
                v16bf a = ld_a(sXCT, NNODE, mt * 16, k0, lane);
                v16bf b = ld_bt(sS, NNODE, k0, nt * 16, lane);   // B = S^T, contiguous from S
                acc = wmma_bf16(a, b, acc);
            }
            const int n  = nt * 16 + ln15;
            const int fb = mt * 16 + half * 8;
            uint4v pk;
#pragma unroll
            for (int i = 0; i < 4; ++i) pk[i] = pack2bf(acc[2 * i], acc[2 * i + 1]);
            *(uint4v*)(sXC + n * Kp + Fp + fb) = pk;
#pragma unroll
            for (int v = 0; v < 8; ++v)
                sXCT[(Fp + fb + v) * NNODE + n] = (__bf16)acc[v];
        }
        __syncthreads();

        // ---- stage2: x2^T = 2*x1^T @ S^T - x0^T  -> XC block2 (packed b128) ----
        for (int t = wv; t < FT * 4; t += NWAVES) {
            int mt = t >> 2, nt = t & 3;
            v8f acc = {};
#pragma unroll
            for (int k0 = 0; k0 < NNODE; k0 += 32) {
                v16bf a = ld_a(sXCT + Fp * NNODE, NNODE, mt * 16, k0, lane);
                v16bf b = ld_bt(sS, NNODE, k0, nt * 16, lane);
                acc = wmma_bf16(a, b, acc);
            }
            const int n  = nt * 16 + ln15;
            const int fb = mt * 16 + half * 8;
            uint4v pk;
#pragma unroll
            for (int i = 0; i < 4; ++i) {
                float v0 = 2.0f * acc[2 * i]     - x0val(n, fb + 2 * i,     phase);
                float v1 = 2.0f * acc[2 * i + 1] - x0val(n, fb + 2 * i + 1, phase);
                pk[i] = pack2bf(v0, v1);
            }
            *(uint4v*)(sXC + n * Kp + 2 * Fp + fb) = pk;
        }
        __syncthreads();

        // ---- output: out^T = W^T @ XC^T ; activation ; GRU epilogue ----
        const int O       = phase ? HD : 2 * HD;
        const __bf16* W   = phase ? Wc : Wg;     // packed column-major = W^T row-major
        const float* bias = phase ? bc : bg;
        const int OT      = O / 16;
        for (int t = wv; t < OT * 4; t += NWAVES) {
            int mt = t >> 2, nt = t & 3;
            v8f acc = {};
            for (int k0 = 0; k0 < Kp; k0 += 32) {
                v16bf a = ld_a(W, Kp, mt * 16, k0, lane);        // global, contiguous
                v16bf b = ld_bt(sXC, Kp, k0, nt * 16, lane);     // LDS, contiguous
                acc = wmma_bf16(a, b, acc);
            }
            const int n  = nt * 16 + ln15;
            const int ob = mt * 16 + half * 8;
            if (phase == 0) {
#pragma unroll
                for (int v = 0; v < 8; ++v) {
                    float pre = acc[v] + bias[ob + v];
                    sRU[n * 2 * HD + ob + v] = 1.0f / (1.0f + __expf(-pre));
                }
            } else {
#pragma unroll
                for (int v = 0; v < 8; ++v) {
                    float c  = tanhf(acc[v] + bias[ob + v]);
                    float u  = sRU[n * 2 * HD + HD + ob + v];
                    float hv = hbuf[n * HD + ob + v];
                    hbuf[n * HD + ob + v] = u * hv + (1.0f - u) * c;
                }
            }
        }
        __syncthreads();
    }
}

// ---- weight prep: fp32 (3F x O) row-major -> bf16 column-major (O x 3*Fp), zero-padded ----
__global__ void pack_w(const float* __restrict__ W, __bf16* __restrict__ dst,
                       int F, int O, int Fp) {
    const int Kp = 3 * Fp;
    const int total = O * Kp;
    for (int idx = blockIdx.x * blockDim.x + threadIdx.x; idx < total;
         idx += gridDim.x * blockDim.x) {
        int o = idx / Kp, kp = idx - o * Kp;
        int i = kp / Fp, f = kp - i * Fp;
        float v = (f < F) ? W[(size_t)(i * F + f) * O + o] : 0.0f;
        dst[idx] = (__bf16)v;
    }
}

// ---- persistent per-batch kernel: full encoder + decoder recurrence ----
__global__ __launch_bounds__(NTHREADS)
void dcrnn_kernel(const float* __restrict__ enc_in,   // (128,64,64,1)
                  const float* __restrict__ dec_in,   // (128,32,64,1)
                  const float* __restrict__ sup,      // (128,64,64,64)
                  const __bf16* e0Wg, const float* e0bg, const __bf16* e0Wc, const float* e0bc,
                  const __bf16* e1Wg, const float* e1bg, const __bf16* e1Wc, const float* e1bc,
                  const __bf16* d0Wg, const float* d0bg, const __bf16* d0Wc, const float* d0bc,
                  const __bf16* d1Wg, const float* d1bg, const __bf16* d1Wc, const float* d1bc,
                  const float* __restrict__ projW, const float* __restrict__ projb,
                  float* __restrict__ out) {
    __shared__ __align__(16) __bf16 sS[NNODE * NNODE];    // 8 KB  adjacency bf16
    __shared__ __align__(16) __bf16 sXC[NNODE * 384];     // 48 KB [x0|x1|x2] node-major
    __shared__ __align__(16) __bf16 sXCT[2 * 128 * NNODE];// 32 KB x0,x1 feat-major
    __shared__ float  sH0[NNODE * HD];                    // 16 KB layer-0 hidden
    __shared__ float  sH1[NNODE * HD];                    // 16 KB layer-1 hidden
    __shared__ float  sRU[NNODE * 2 * HD];                // 32 KB gates r|u
    __shared__ float  sXV[NNODE];

    const int b   = blockIdx.x;
    const int tid = threadIdx.x;

    for (int i = tid; i < NNODE * HD; i += NTHREADS) { sH0[i] = 0.0f; sH1[i] = 0.0f; }
    __syncthreads();

    // ---------------- encoder: 64 steps, two stacked cells ----------------
    for (int t = 0; t < 64; ++t) {
        const float* Sp = sup + ((size_t)(b * 64 + t)) * (NNODE * NNODE);
        for (int i = tid; i < NNODE * NNODE; i += NTHREADS) sS[i] = (__bf16)Sp[i];
        if (tid < NNODE) sXV[tid] = enc_in[(size_t)(b * 64 + t) * NNODE + tid];
        // prefetch next step's 16 KB S tile (256 threads x 64B lines) -> global_prefetch_b8;
        // hides the only critical-path global load behind this step's WMMA work.
        if (t + 1 < 64) {
            const char* Pn = (const char*)(sup + ((size_t)(b * 64 + t + 1)) * (NNODE * NNODE));
            __builtin_prefetch(Pn + (size_t)tid * 64, 0, 1);
        }
        __syncthreads();

        dcgru_cell<1>(sS, sXC, sXCT, sRU, sXV, sH0, nullptr, e0Wg, e0bg, e0Wc, e0bc);
        dcgru_cell<64>(sS, sXC, sXCT, sRU, sXV, sH1, sH0,    e1Wg, e1bg, e1Wc, e1bc);
    }

    // ---------------- decoder: fixed S = supports[:, -1] ----------------
    {
        const float* Sp = sup + ((size_t)(b * 64 + 63)) * (NNODE * NNODE);
        for (int i = tid; i < NNODE * NNODE; i += NTHREADS) sS[i] = (__bf16)Sp[i];
        if (tid < NNODE) sXV[tid] = dec_in[(size_t)(b * 32 + 0) * NNODE + tid];
        __syncthreads();
    }
    for (int t = 0; t < 32; ++t) {
        dcgru_cell<1>(sS, sXC, sXCT, sRU, sXV, sH0, nullptr, d0Wg, d0bg, d0Wc, d0bc);
        dcgru_cell<64>(sS, sXC, sXCT, sRU, sXV, sH1, sH0,    d1Wg, d1bg, d1Wc, d1bc);

        // projection (N,H) @ (H,1) + b ; feed back as next input
        if (tid < NNODE) {
            float acc = projb[0];
#pragma unroll 8
            for (int j = 0; j < HD; ++j) acc += sH1[tid * HD + j] * projW[j];
            out[(size_t)(b * 32 + t) * NNODE + tid] = acc;
            sXV[tid] = acc;
        }
        __syncthreads();
    }
}

extern "C" void kernel_launch(void* const* d_in, const int* in_sizes, int n_in,
                              void* d_out, int out_size, void* d_ws, size_t ws_size,
                              hipStream_t stream) {
    (void)in_sizes; (void)n_in; (void)out_size; (void)ws_size;
    const float* enc_in = (const float*)d_in[0];
    const float* dec_in = (const float*)d_in[1];
    const float* sup    = (const float*)d_in[2];
    const float* e0Wg = (const float*)d_in[3];  const float* e0bg = (const float*)d_in[4];
    const float* e0Wc = (const float*)d_in[5];  const float* e0bc = (const float*)d_in[6];
    const float* e1Wg = (const float*)d_in[7];  const float* e1bg = (const float*)d_in[8];
    const float* e1Wc = (const float*)d_in[9];  const float* e1bc = (const float*)d_in[10];
    const float* d0Wg = (const float*)d_in[11]; const float* d0bg = (const float*)d_in[12];
    const float* d0Wc = (const float*)d_in[13]; const float* d0bc = (const float*)d_in[14];
    const float* d1Wg = (const float*)d_in[15]; const float* d1bg = (const float*)d_in[16];
    const float* d1Wc = (const float*)d_in[17]; const float* d1bc = (const float*)d_in[18];
    const float* projW = (const float*)d_in[19];
    const float* projb = (const float*)d_in[20];

    // workspace: packed bf16 weights, column-major, K padded (Fp=96 for din=1, 128 for din=64)
    __bf16* ws = (__bf16*)d_ws;
    size_t off = 0;
    __bf16* pe0Wg = ws + off; off += (size_t)128 * 288;
    __bf16* pe0Wc = ws + off; off += (size_t)64  * 288;
    __bf16* pe1Wg = ws + off; off += (size_t)128 * 384;
    __bf16* pe1Wc = ws + off; off += (size_t)64  * 384;
    __bf16* pd0Wg = ws + off; off += (size_t)128 * 288;
    __bf16* pd0Wc = ws + off; off += (size_t)64  * 288;
    __bf16* pd1Wg = ws + off; off += (size_t)128 * 384;
    __bf16* pd1Wc = ws + off; off += (size_t)64  * 384;

    pack_w<<<64, 256, 0, stream>>>(e0Wg, pe0Wg, 65, 128, 96);
    pack_w<<<64, 256, 0, stream>>>(e0Wc, pe0Wc, 65, 64, 96);
    pack_w<<<64, 256, 0, stream>>>(e1Wg, pe1Wg, 128, 128, 128);
    pack_w<<<64, 256, 0, stream>>>(e1Wc, pe1Wc, 128, 64, 128);
    pack_w<<<64, 256, 0, stream>>>(d0Wg, pd0Wg, 65, 128, 96);
    pack_w<<<64, 256, 0, stream>>>(d0Wc, pd0Wc, 65, 64, 96);
    pack_w<<<64, 256, 0, stream>>>(d1Wg, pd1Wg, 128, 128, 128);
    pack_w<<<64, 256, 0, stream>>>(d1Wc, pd1Wc, 128, 64, 128);

    dcrnn_kernel<<<128, NTHREADS, 0, stream>>>(
        enc_in, dec_in, sup,
        pe0Wg, e0bg, pe0Wc, e0bc,
        pe1Wg, e1bg, pe1Wc, e1bc,
        pd0Wg, d0bg, pd0Wc, d0bc,
        pd1Wg, d1bg, pd1Wc, d1bc,
        projW, projb, (float*)d_out);
}
